// UGCABlock_67568425501332
// MI455X (gfx1250) — matrix operand order repaired and checked by reading, not compiled, not run on hardware
//
#include <hip/hip_runtime.h>
#include <hip/hip_bf16.h>
#include <math.h>

#define B_ROWS 16384
#define DDIM   1024

typedef __bf16 bf16_t;
typedef __attribute__((ext_vector_type(16))) __bf16 v16bf;
typedef __attribute__((ext_vector_type(8)))  __bf16 v8bf;
typedef __attribute__((ext_vector_type(8)))  float  v8f;

union V16 { v16bf v; v8bf h[2]; };

__device__ inline float wave_sum(float v) {
    #pragma unroll
    for (int m = 16; m > 0; m >>= 1) v += __shfl_xor(v, m, 32);
    return v;
}

__device__ inline float softplus_f(float z) {
    return fmaxf(z, 0.0f) + log1pf(expf(-fabsf(z)));
}

// async global->LDS copy of 16 bytes (gfx1250 GLOBAL_LOAD_ASYNC_TO_LDS_B128,
// tracked with ASYNCcnt; LDS dest offset = low 32 bits of the flat pointer).
__device__ inline void async_copy_b128(const void* gsrc, void* ldst) {
    unsigned loff = (unsigned)(uintptr_t)ldst;
    asm volatile("global_load_async_to_lds_b128 %0, %1, off"
                 :: "v"(loff), "v"(gsrc) : "memory");
}
__device__ inline void async_wait_all() {
    asm volatile("s_wait_asynccnt 0" ::: "memory");
}

// ---------------------------------------------------------------------------
// Kernel 0: convert Wq/Wk/Wv (fp32, D x D) -> packed bf16 Wb[which][D][D].
// ---------------------------------------------------------------------------
__global__ __launch_bounds__(256)
void cvt_w_kernel(const float* __restrict__ Wq, const float* __restrict__ Wk,
                  const float* __restrict__ Wv, bf16_t* __restrict__ Wb)
{
    const int which = blockIdx.y;
    const float* W = (which == 0) ? Wq : ((which == 1) ? Wk : Wv);
    bf16_t* o = Wb + (size_t)which * DDIM * DDIM;
    const long i = ((long)blockIdx.x * 256 + threadIdx.x) * 8;   // grid.x = D*D/2048
    float4 x0 = *(const float4*)(W + i);
    float4 x1 = *(const float4*)(W + i + 4);
    v8bf p;
    p[0] = (bf16_t)x0.x; p[1] = (bf16_t)x0.y; p[2] = (bf16_t)x0.z; p[3] = (bf16_t)x0.w;
    p[4] = (bf16_t)x1.x; p[5] = (bf16_t)x1.y; p[6] = (bf16_t)x1.z; p[7] = (bf16_t)x1.w;
    *(v8bf*)(o + i) = p;
}

// ---------------------------------------------------------------------------
// Kernel 1: per-row reliability + gate half-dots + bf16 copy of H.
// Streams each h row exactly once; one wave per row, 8 rows per block.
// ---------------------------------------------------------------------------
__global__ __launch_bounds__(256)
void rel_gate_kernel(const float* __restrict__ h0, const float* __restrict__ h1,
                     const float* __restrict__ h2, const float* __restrict__ h3,
                     const float* __restrict__ rw0, const float* __restrict__ rb0,
                     const float* __restrict__ rw1, const float* __restrict__ rb1,
                     const float* __restrict__ rw2, const float* __restrict__ rb2,
                     const float* __restrict__ rw3, const float* __restrict__ rb3,
                     const float* __restrict__ gate_w,
                     bf16_t* __restrict__ Hb,
                     float* __restrict__ r_out, float* __restrict__ ga_out,
                     float* __restrict__ gb_out)
{
    const int lane = threadIdx.x & 31;
    const int wave = threadIdx.x >> 5;
    // src is block-uniform (8 | 16384): keep it scalar so loads use s[base]+voff
    const int  src = (int)(((long)blockIdx.x * 8) >> 14);
    const long row = (long)blockIdx.x * 8 + wave;
    const long b   = row - (long)src * B_ROWS;

    const float* hsel = (src == 0) ? h0 : ((src == 1) ? h1 : ((src == 2) ? h2 : h3));
    const float* rwsel = (src == 0) ? rw0 : ((src == 1) ? rw1 : ((src == 2) ? rw2 : rw3));
    const float* rbsel = (src == 0) ? rb0 : ((src == 1) ? rb1 : ((src == 2) ? rb2 : rb3));

    const float* hp = hsel + b * DDIM + lane * 32;
    const float* w0 = rwsel + 0 * DDIM + lane * 32;
    const float* w1 = rwsel + 1 * DDIM + lane * 32;
    const float* gA = gate_w + lane * 32;
    const float* gB = gate_w + DDIM + lane * 32;
    bf16_t* hbp = Hb + row * DDIM + lane * 32;

    float d0 = 0.f, d1 = 0.f, dA = 0.f, dB = 0.f;
    #pragma unroll
    for (int u = 0; u < 32; u += 8) {
        float4 xa = *(const float4*)(hp + u);
        float4 xb = *(const float4*)(hp + u + 4);
        float4 a0 = *(const float4*)(w0 + u);
        float4 a1 = *(const float4*)(w0 + u + 4);
        float4 b0 = *(const float4*)(w1 + u);
        float4 b1 = *(const float4*)(w1 + u + 4);
        float4 c0 = *(const float4*)(gA + u);
        float4 c1 = *(const float4*)(gA + u + 4);
        float4 e0 = *(const float4*)(gB + u);
        float4 e1 = *(const float4*)(gB + u + 4);
        d0 += xa.x*a0.x + xa.y*a0.y + xa.z*a0.z + xa.w*a0.w
            + xb.x*a1.x + xb.y*a1.y + xb.z*a1.z + xb.w*a1.w;
        d1 += xa.x*b0.x + xa.y*b0.y + xa.z*b0.z + xa.w*b0.w
            + xb.x*b1.x + xb.y*b1.y + xb.z*b1.z + xb.w*b1.w;
        dA += xa.x*c0.x + xa.y*c0.y + xa.z*c0.z + xa.w*c0.w
            + xb.x*c1.x + xb.y*c1.y + xb.z*c1.z + xb.w*c1.w;
        dB += xa.x*e0.x + xa.y*e0.y + xa.z*e0.z + xa.w*e0.w
            + xb.x*e1.x + xb.y*e1.y + xb.z*e1.z + xb.w*e1.w;
        v8bf p;
        p[0] = (bf16_t)xa.x; p[1] = (bf16_t)xa.y; p[2] = (bf16_t)xa.z; p[3] = (bf16_t)xa.w;
        p[4] = (bf16_t)xb.x; p[5] = (bf16_t)xb.y; p[6] = (bf16_t)xb.z; p[7] = (bf16_t)xb.w;
        *(v8bf*)(hbp + u) = p;
    }
    d0 = wave_sum(d0); d1 = wave_sum(d1);
    dA = wave_sum(dA); dB = wave_sum(dB);

    if (lane == 0) {
        float e1 = softplus_f(d0 + rbsel[0]);
        float e2 = softplus_f(d1 + rbsel[1]);
        float S  = e1 + e2 + 2.0f;
        float r  = 1.0f - 2.0f / (S + 1e-8f);
        r = fminf(fmaxf(r, 0.0f), 1.0f);
        r_out[row]  = r;
        ga_out[row] = dA;
        gb_out[row] = dB;
    }
}

// ---------------------------------------------------------------------------
// Kernel 2: Q/K/V projection GEMM, all-bf16 inputs, WMMA + async-LDS path.
//   Out[row, n] = sum_k Hb[row, k] * Wb[which][n, k]
// Grid: x = which*8 + ntile (24 launch-adjacent blocks share one 256 KB A
// stripe in L2), y = mtile (512). WG tile 128x128; 8 waves of 64x32 (4x2
// frags, K-step 32) -> ~130 live VGPRs, no VGPR-MSB spill-copy traffic.
// Tiles staged into double-buffered padded LDS exclusively with
// GLOBAL_LOAD_ASYNC_TO_LDS_B128 (ASYNCcnt), zero staging VALU.
// ---------------------------------------------------------------------------
__global__ __launch_bounds__(256)
void qkv_gemm_kernel(const bf16_t* __restrict__ Hb, const bf16_t* __restrict__ Wb,
                     bf16_t* __restrict__ Qw, bf16_t* __restrict__ Kw,
                     bf16_t* __restrict__ Vw)
{
    constexpr int TK  = 32;   // K per step
    constexpr int LST = 40;   // padded LDS row stride (bf16) -> conflict-free frags
    __shared__ bf16_t lds_a[2][128 * LST];
    __shared__ bf16_t lds_b[2][128 * LST];

    const int tid   = threadIdx.x;
    const int which = blockIdx.x >> 3;    // 0:Q 1:K 2:V
    const int ntile = blockIdx.x & 7;     // 128-wide N stripes
    const int mtile = blockIdx.y;         // 0..511

    const bf16_t* W = Wb + (size_t)which * DDIM * DDIM;
    bf16_t* Out = (which == 0) ? Qw : ((which == 1) ? Kw : Vw);

    // staging: 2 threads per row, 16 bf16 (2 x async b128) each, for A and B
    const int srow = tid >> 1;
    const int scol = (tid & 1) * 16;
    const bf16_t* agp = Hb + ((long)mtile * 128 + srow) * DDIM + scol;
    const bf16_t* bgp = W + ((long)ntile * 128 + srow) * DDIM + scol;

    auto stage = [&](int buf, int kt) {
        const bf16_t* ga = agp + kt * TK;
        bf16_t* la = &lds_a[buf][srow * LST + scol];
        async_copy_b128(ga,     la);
        async_copy_b128(ga + 8, la + 8);
        const bf16_t* gb = bgp + kt * TK;
        bf16_t* lb = &lds_b[buf][srow * LST + scol];
        async_copy_b128(gb,     lb);
        async_copy_b128(gb + 8, lb + 8);
    };

    // per-wave compute geometry: 64(M) x 32(N) tile
    const int lane = tid & 31;
    const int wave = tid >> 5;
    const int wm   = (wave >> 2) * 64;   // 0/64
    const int wn   = (wave & 3) * 32;    // 0/32/64/96
    const int frow = lane & 15;          // M for A frag, N for B frag
    const int fsel = lane >> 4;          // lane-half select per ISA layout

    v8f acc[4][2];
    #pragma unroll
    for (int i = 0; i < 4; i++)
        #pragma unroll
        for (int j = 0; j < 2; j++)
            acc[i][j] = (v8f){0.f,0.f,0.f,0.f,0.f,0.f,0.f,0.f};

    constexpr int NT = DDIM / TK;   // 32 K-steps
    stage(0, 0);
    async_wait_all();
    __syncthreads();

    for (int kt = 0; kt < NT; ++kt) {
        if (kt + 1 < NT) stage((kt + 1) & 1, kt + 1);

        const bf16_t* la = lds_a[kt & 1];
        const bf16_t* lb = lds_b[kt & 1];

        // A 16x32 frag: lanes 0-15 hold K 0..7 & 16..23; lanes 16-31 K 8..15 & 24..31
        V16 afrag[4];
        #pragma unroll
        for (int i = 0; i < 4; i++) {
            const bf16_t* p = &la[(wm + 16 * i + frow) * LST + fsel * 8];
            afrag[i].h[0] = *(const v8bf*)(p);
            afrag[i].h[1] = *(const v8bf*)(p + 16);
        }
        // B 32x16 frag: lanes 0-15 hold K 0..7 & 16..23; lanes 16-31 K 8..15 & 24..31 (N = lane&15)
        V16 bfrag[2];
        #pragma unroll
        for (int j = 0; j < 2; j++) {
            const bf16_t* p = &lb[(wn + 16 * j + frow) * LST + fsel * 8];
            bfrag[j].h[0] = *(const v8bf*)(p);
            bfrag[j].h[1] = *(const v8bf*)(p + 16);
        }

        #pragma unroll
        for (int i = 0; i < 4; i++)
            #pragma unroll
            for (int j = 0; j < 2; j++)
                acc[i][j] = __builtin_amdgcn_wmma_f32_16x16x32_bf16(
                    false, afrag[i].v, false, bfrag[j].v,
                    (short)0, acc[i][j], false, false);

        async_wait_all();
        __syncthreads();
    }

    // C/D layout: VGPR e -> row = e + 8*(lane>>4), col = lane&15
    #pragma unroll
    for (int i = 0; i < 4; i++) {
        #pragma unroll
        for (int j = 0; j < 2; j++) {
            #pragma unroll
            for (int e = 0; e < 8; e++) {
                long row = (long)mtile * 128 + wm + 16 * i + e + 8 * fsel;
                int  col = ntile * 128 + wn + 16 * j + frow;
                Out[row * DDIM + col] = (bf16_t)acc[i][j][e];
            }
        }
    }
}

// ---------------------------------------------------------------------------
// Kernel 3: fused attend epilogue, one wave per (call, row).
//   call c: qs = c, keys = (c<2) ? {esm,poc} : {mc,cb}
// ---------------------------------------------------------------------------
__global__ __launch_bounds__(256)
void attend_kernel(const float* __restrict__ h0, const float* __restrict__ h1,
                   const float* __restrict__ h2, const float* __restrict__ h3,
                   const bf16_t* __restrict__ Qw, const bf16_t* __restrict__ Kw,
                   const bf16_t* __restrict__ Vw,
                   const float* __restrict__ rbuf, const float* __restrict__ gabuf,
                   const float* __restrict__ gbbuf,
                   const float* __restrict__ gate_b,
                   const float* __restrict__ ln_g, const float* __restrict__ ln_b,
                   float* __restrict__ out)
{
    const int  lane = threadIdx.x & 31;
    const int  wave = threadIdx.x >> 5;
    // call index is block-uniform (8 | 16384): keep scalar
    const int  c    = (int)(((long)blockIdx.x * 8) >> 14);
    const long idx  = (long)blockIdx.x * 8 + wave;
    const long b    = idx - (long)c * B_ROWS;
    const int  k1   = (c < 2) ? 2 : 0;
    const int  k2   = (c < 2) ? 3 : 1;

    const float* hq = (c == 0) ? h0 : ((c == 1) ? h1 : ((c == 2) ? h2 : h3));

    // ---- logits: q . k over D, 32 elems per lane ----
    const bf16_t* qp  = Qw + ((long)c  * B_ROWS + b) * DDIM + lane * 32;
    const bf16_t* kp1 = Kw + ((long)k1 * B_ROWS + b) * DDIM + lane * 32;
    const bf16_t* kp2 = Kw + ((long)k2 * B_ROWS + b) * DDIM + lane * 32;
    float d1 = 0.f, d2 = 0.f;
    #pragma unroll
    for (int u = 0; u < 32; u += 8) {
        v8bf q  = *(const v8bf*)(qp  + u);
        v8bf a  = *(const v8bf*)(kp1 + u);
        v8bf bb = *(const v8bf*)(kp2 + u);
        #pragma unroll
        for (int j = 0; j < 8; j++) {
            float qf = (float)q[j];
            d1 += qf * (float)a[j];
            d2 += qf * (float)bb[j];
        }
    }
    d1 = wave_sum(d1);
    d2 = wave_sum(d2);
    const float inv_sqrt_d = 1.0f / 32.0f;   // 1/sqrt(1024)
    float l1 = d1 * inv_sqrt_d, l2 = d2 * inv_sqrt_d;

    // ---- gates (scalars) ----
    float rq = rbuf[(long)c  * B_ROWS + b];
    float r1 = rbuf[(long)k1 * B_ROWS + b];
    float r2 = rbuf[(long)k2 * B_ROWS + b];
    float aq = gabuf[(long)c * B_ROWS + b];
    float gbias = gate_b[0];
    float g1 = 1.0f / (1.0f + expf(-(aq + gbbuf[(long)k1 * B_ROWS + b] + gbias)));
    float g2 = 1.0f / (1.0f + expf(-(aq + gbbuf[(long)k2 * B_ROWS + b] + gbias)));
    float G1 = g1 * rq * r1, G2 = g2 * rq * r2;

    // ---- 2-way softmax ----
    float z1 = l1 * G1, z2 = l2 * G2;
    float zm = fmaxf(z1, z2);
    float e1 = expf(z1 - zm), e2 = expf(z2 - zm);
    float s  = e1 + e2;
    float a1 = e1 / s, a2 = e2 / s;

    // ---- ctx + residual, layernorm ----
    const bf16_t* vp1 = Vw + ((long)k1 * B_ROWS + b) * DDIM + lane * 32;
    const bf16_t* vp2 = Vw + ((long)k2 * B_ROWS + b) * DDIM + lane * 32;
    const float*  hp  = hq + b * DDIM + lane * 32;

    float x[32];
    float sum = 0.f, sq = 0.f;
    #pragma unroll
    for (int u = 0; u < 32; u += 8) {
        v8bf va = *(const v8bf*)(vp1 + u);
        v8bf vb = *(const v8bf*)(vp2 + u);
        float4 hA = *(const float4*)(hp + u);
        float4 hB = *(const float4*)(hp + u + 4);
        float hv[8] = {hA.x, hA.y, hA.z, hA.w, hB.x, hB.y, hB.z, hB.w};
        #pragma unroll
        for (int j = 0; j < 8; j++) {
            float xv = hv[j] + a1 * (float)va[j] + a2 * (float)vb[j];
            x[u + j] = xv;
            sum += xv;
            sq  += xv * xv;
        }
    }
    sum = wave_sum(sum);
    sq  = wave_sum(sq);
    const float invD = 1.0f / (float)DDIM;
    float mean = sum * invD;
    float var  = sq * invD - mean * mean;
    float rstd = rsqrtf(var + 1e-5f);

    float*       op = out  + idx * (DDIM + 1) + lane * 32;
    const float* lg = ln_g + lane * 32;
    const float* lb = ln_b + lane * 32;
    #pragma unroll
    for (int u = 0; u < 32; u++)
        op[u] = (x[u] - mean) * rstd * lg[u] + lb[u];

    if (lane == 0)
        out[idx * (DDIM + 1) + DDIM] = rq;   // r_stack column
}

// ---------------------------------------------------------------------------
extern "C" void kernel_launch(void* const* d_in, const int* in_sizes, int n_in,
                              void* d_out, int out_size, void* d_ws, size_t ws_size,
                              hipStream_t stream)
{
    (void)in_sizes; (void)n_in; (void)out_size; (void)ws_size;
    const float* h_mc  = (const float*)d_in[0];
    const float* h_cb  = (const float*)d_in[1];
    const float* h_esm = (const float*)d_in[2];
    const float* h_poc = (const float*)d_in[3];
    const float* Wq    = (const float*)d_in[4];
    const float* Wk    = (const float*)d_in[5];
    const float* Wv    = (const float*)d_in[6];
    const float* gate_w = (const float*)d_in[7];
    const float* gate_b = (const float*)d_in[8];
    const float* ln_g   = (const float*)d_in[9];
    const float* ln_b   = (const float*)d_in[10];
    const float* rw_mc  = (const float*)d_in[11];
    const float* rb_mc  = (const float*)d_in[12];
    const float* rw_cb  = (const float*)d_in[13];
    const float* rb_cb  = (const float*)d_in[14];
    const float* rw_esm = (const float*)d_in[15];
    const float* rb_esm = (const float*)d_in[16];
    const float* rw_poc = (const float*)d_in[17];
    const float* rb_poc = (const float*)d_in[18];

    float* out = (float*)d_out;

    char* ws = (char*)d_ws;
    const size_t HSZ = (size_t)4 * B_ROWS * DDIM * sizeof(bf16_t);   // 128 MB
    const size_t WSZ = (size_t)3 * DDIM * DDIM * sizeof(bf16_t);     // 6 MB
    bf16_t* Hb = (bf16_t*)(ws);
    bf16_t* Wb = (bf16_t*)(ws + HSZ);
    bf16_t* Qw = (bf16_t*)(ws + HSZ + WSZ);
    bf16_t* Kw = (bf16_t*)(ws + 2 * HSZ + WSZ);
    bf16_t* Vw = (bf16_t*)(ws + 3 * HSZ + WSZ);
    float*  rbuf  = (float*)(ws + 4 * HSZ + WSZ);
    float*  gabuf = rbuf  + (size_t)4 * B_ROWS;
    float*  gbbuf = gabuf + (size_t)4 * B_ROWS;

    // 0) weights -> bf16 (12 MB, trivial)
    dim3 wgrid(DDIM * DDIM / 2048, 3);
    cvt_w_kernel<<<wgrid, 256, 0, stream>>>(Wq, Wk, Wv, Wb);

    // 1) reliability + gate scalars + packed bf16 H (single pass over H)
    rel_gate_kernel<<<(4 * B_ROWS) / 8, 256, 0, stream>>>(
        h_mc, h_cb, h_esm, h_poc,
        rw_mc, rb_mc, rw_cb, rb_cb, rw_esm, rb_esm, rw_poc, rb_poc,
        gate_w, Hb, rbuf, gabuf, gbbuf);

    // 2) 12 projection GEMMs on the WMMA bf16 + async-LDS path
    dim3 gemm_grid(24, (4 * B_ROWS) / 128);
    qkv_gemm_kernel<<<gemm_grid, 256, 0, stream>>>(Hb, Wb, Qw, Kw, Vw);

    // 3) fused attend + layernorm epilogue (wave per (call,row))
    attend_kernel<<<(4 * B_ROWS) / 8, 256, 0, stream>>>(
        h_mc, h_cb, h_esm, h_poc, Qw, Kw, Vw,
        rbuf, gabuf, gbbuf, gate_b, ln_g, ln_b, out);
}
